// NSE_37787122270951
// MI455X (gfx1250) — matrix operand order repaired and checked by reading, not compiled, not run on hardware
//
#include <hip/hip_runtime.h>
#include <hip/hip_bf16.h>

#define T_STEPS 64
#define B 32
#define N 512
#define D 512
#define EPS 1e-8f
#define NEG_INF -3.40282347e38f

typedef __attribute__((ext_vector_type(16))) __bf16 v16bf;
typedef __attribute__((ext_vector_type(8)))  float  v8f;

__device__ __forceinline__ float sigmoidf_(float x) { return 1.0f / (1.0f + __expf(-x)); }

// ================= packed-tile layouts (wave32 WMMA bf16 16x16x32) =============
// A-tile (16x32 MxK): per tile, 32 lanes x 16 bf16. lane: m=lane&15, hi=lane>>4.
//   element e (v=e>>1, rem=e&1) -> k = hi*8 + (v&3)*2 + (v&4)*4 + rem.
// B-tile (32x16 KxN): lane: n=lane&15, hi=lane>>4; element e -> k=hi*8 + (e&7) + (e>>3)*16.
// Tiles stored contiguously: tile*512 + lane*16 + e.

__device__ __forceinline__ v16bf load_tile(const __bf16* __restrict__ base, int tileIdx) {
  const int lane = threadIdx.x & 31;
  return *(const v16bf*)(base + ((size_t)tileIdx << 9) + (lane << 4));
}

// Inverse map: element (row b, col d) -> flat index in packed A layout
// (tiles ordered [b-tile][k-tile], k-tiles = D/32).
__device__ __forceinline__ int packA_index(int b, int d) {
  const int tb  = b >> 4, m = b & 15;
  const int kt  = d >> 5, kin = d & 31;
  const int hi  = (kin >> 3) & 1;
  const int v   = ((kin >> 4) << 2) + ((kin >> 1) & 3);
  const int rem = kin & 1;
  const int lane = (hi << 4) + m;
  const int e    = (v << 1) + rem;
  return ((((tb * (D >> 5)) + kt) * 32 + lane) << 4) + e;
}

// ---- one-time pack: W [R x K] row-major fp32 -> packed bf16 B-tiles -----------
__global__ __launch_bounds__(256) void pack_w(
    const float* __restrict__ W, __bf16* __restrict__ out, int R, int K)
{
  const int i = blockIdx.x * 256 + threadIdx.x;
  if (i >= R * K) return;
  const int e    = i & 15;
  const int lane = (i >> 4) & 31;
  const int tile = i >> 9;
  const int ktiles = K >> 5;
  const int kt = tile % ktiles;
  const int jt = tile / ktiles;
  const int n  = lane & 15;
  const int k  = ((lane >> 4) << 3) + (e & 7) + ((e >> 3) << 4);
  out[i] = (__bf16)W[(size_t)(jt * 16 + n) * K + (kt << 5) + k];
}

// ---- one-time pack: batched activations [pages x B x D] -> packed A-tiles -----
__global__ __launch_bounds__(256) void pack_a_batch(
    const float* __restrict__ src, __bf16* __restrict__ out, int total)
{
  const int i = blockIdx.x * 256 + threadIdx.x;
  if (i >= total) return;
  const int page = i >> 14;          // B*D = 16384 elements per page
  const int il   = i & 16383;
  const int e    = il & 15;
  const int lane = (il >> 4) & 31;
  const int tile = il >> 9;
  const int kt = tile & 15;          // D/32 = 16 k-tiles
  const int tb = tile >> 4;
  const int m = lane & 15, hi = lane >> 4;
  const int v = e >> 1, rem = e & 1;
  const int k = (hi << 3) + ((v & 3) << 1) + ((v & 4) << 2) + rem;
  out[i] = (__bf16)src[(size_t)page * (B * D) + (size_t)(tb * 16 + m) * D + (kt << 5) + k];
}

// ---- fused LSTM cell: g = x@W_ih^T + h@W_hh^T + b, gates, packed-h output -----
__global__ __launch_bounds__(32) void nse_lstm_gates(
    const __bf16* __restrict__ pX, const __bf16* __restrict__ pH,
    const float* __restrict__ c_in,
    const __bf16* __restrict__ pWih, const __bf16* __restrict__ pWhh,
    const float* __restrict__ bias,
    float* __restrict__ h_out, float* __restrict__ c_out,
    float* __restrict__ h_out2, __bf16* __restrict__ pHout)
{
  const int nt = blockIdx.x;       // d-tile 0..31
  const int tb = blockIdx.y;       // batch-tile 0..1
  const int n0 = nt << 4, b0 = tb << 4;
  const int KT = D >> 5;           // 16 k-tiles
  v8f zero = {};
  v8f acc[4] = {zero, zero, zero, zero};

  for (int kt = 0; kt < KT; ++kt) {
    v16bf ax = load_tile(pX, tb * KT + kt);
    v16bf ah = load_tile(pH, tb * KT + kt);
#pragma unroll
    for (int q = 0; q < 4; ++q) {
      const int jt = q * (D >> 4) + nt;  // row-tile within [4D/16]
      v16bf bi = load_tile(pWih, jt * KT + kt);
      acc[q] = __builtin_amdgcn_wmma_f32_16x16x32_bf16(
          false, ax, false, bi, (short)0, acc[q], false, false);
      v16bf bh = load_tile(pWhh, jt * KT + kt);
      acc[q] = __builtin_amdgcn_wmma_f32_16x16x32_bf16(
          false, ah, false, bh, (short)0, acc[q], false, false);
    }
  }

  const int lane = threadIdx.x & 31;
  const int nn = lane & 15;
  const int hi = lane >> 4;
  const float bi_ = bias[0 * D + n0 + nn];
  const float bf_ = bias[1 * D + n0 + nn];
  const float bg_ = bias[2 * D + n0 + nn];
  const float bo_ = bias[3 * D + n0 + nn];
#pragma unroll
  for (int r = 0; r < 8; ++r) {
    const int b = b0 + r + (hi << 3);
    const int d = n0 + nn;
    const float gi = sigmoidf_(acc[0][r] + bi_);
    const float gf = sigmoidf_(acc[1][r] + bf_);
    const float gg = tanhf(acc[2][r] + bg_);
    const float go = sigmoidf_(acc[3][r] + bo_);
    const float c2 = gf * c_in[(size_t)b * D + d] + gi * gg;
    const float h2 = go * tanhf(c2);
    c_out[(size_t)b * D + d] = c2;
    h_out[(size_t)b * D + d] = h2;
    if (h_out2) h_out2[(size_t)b * D + d] = h2;
    pHout[packA_index(b, d)] = (__bf16)h2;
  }
}

// ---- per-(b,n): dot(hr, M[b,n,:]) and ||M[b,n,:]|| ----------------------------
__global__ __launch_bounds__(256) void nse_sim(
    const float* __restrict__ hr, const float* __restrict__ M,
    float* __restrict__ numv, float* __restrict__ nMv)
{
  const int lane = threadIdx.x & 31;
  const int row = blockIdx.x * 8 + (threadIdx.x >> 5);  // row = b*N + n
  const int b = row >> 9;                                // N = 512
  const float* Mr = M + (size_t)row * D;
  const float* hb = hr + (size_t)b * D;
  float s_num = 0.f, s_nm = 0.f;
  for (int d = lane; d < D; d += 32) {
    const float mv = Mr[d];
    s_num += hb[d] * mv;
    s_nm  += mv * mv;
  }
#pragma unroll
  for (int off = 16; off > 0; off >>= 1) {
    s_num += __shfl_xor(s_num, off, 32);
    s_nm  += __shfl_xor(s_nm, off, 32);
  }
  if (lane == 0) { numv[row] = s_num; nMv[row] = fmaxf(sqrtf(s_nm), EPS); }
}

// ---- per-batch masked softmax over N=512 similarities -> z --------------------
__global__ __launch_bounds__(256) void nse_softmax_z(
    const float* __restrict__ hr, const float* __restrict__ numv,
    const float* __restrict__ nMv, const unsigned char* __restrict__ mask,
    float* __restrict__ z)
{
  __shared__ float red[256];
  const int b = blockIdx.x;
  const int tid = threadIdx.x;
  float s = 0.f;
  for (int d = tid; d < D; d += 256) { const float v = hr[(size_t)b * D + d]; s += v * v; }
  red[tid] = s; __syncthreads();
  for (int off = 128; off > 0; off >>= 1) { if (tid < off) red[tid] += red[tid + off]; __syncthreads(); }
  const float nh = fmaxf(sqrtf(red[0]), EPS);
  __syncthreads();

  float sim[2];
#pragma unroll
  for (int j = 0; j < 2; ++j) {
    const int idx = b * N + tid + j * 256;
    sim[j] = mask[idx] ? NEG_INF : numv[idx] / (nh * nMv[idx]);
  }
  red[tid] = fmaxf(sim[0], sim[1]); __syncthreads();
  for (int off = 128; off > 0; off >>= 1) { if (tid < off) red[tid] = fmaxf(red[tid], red[tid + off]); __syncthreads(); }
  const float mx = red[0];
  __syncthreads();
  const float e0 = __expf(sim[0] - mx), e1 = __expf(sim[1] - mx);
  red[tid] = e0 + e1; __syncthreads();
  for (int off = 128; off > 0; off >>= 1) { if (tid < off) red[tid] += red[tid + off]; __syncthreads(); }
  const float inv = 1.0f / red[0];
  z[b * N + tid]       = e0 * inv;
  z[b * N + tid + 256] = e1 * inv;
}

// ---- memory read m[b,d] = sum_n z[b,n]*M[b,n,d]; output packed bf16 A-tiles ---
__global__ __launch_bounds__(256) void nse_read(
    const float* __restrict__ z, const float* __restrict__ M,
    __bf16* __restrict__ pM)
{
  const int b = blockIdx.y;
  const int d = blockIdx.x * 256 + threadIdx.x;
  const float* zb = z + (size_t)b * N;
  const float* Mb = M + (size_t)b * N * D + d;
  float acc = 0.f;
  for (int n = 0; n < N; ++n) acc += zb[n] * Mb[(size_t)n * D];
  pM[packA_index(b, d)] = (__bf16)acc;
}

// ---- comp logits = concat(hr, m) @ W_comp^T + b_comp (WMMA, packed operands) --
__global__ __launch_bounds__(32) void nse_comp_gemm(
    const __bf16* __restrict__ pHr, const __bf16* __restrict__ pM,
    const __bf16* __restrict__ pWc, const float* __restrict__ bc,
    float* __restrict__ logits)
{
  const int nt = blockIdx.x;
  const int tb = blockIdx.y;
  const int n0 = nt << 4, b0 = tb << 4;
  const int KT  = D >> 5;         // 16
  const int KT2 = (2 * D) >> 5;   // 32
  v8f acc = {};
  for (int kt = 0; kt < KT2; ++kt) {
    v16bf a = (kt < KT) ? load_tile(pHr, tb * KT + kt)
                        : load_tile(pM,  tb * KT + (kt - KT));
    v16bf bt = load_tile(pWc, nt * KT2 + kt);
    acc = __builtin_amdgcn_wmma_f32_16x16x32_bf16(
        false, a, false, bt, (short)0, acc, false, false);
  }
  const int lane = threadIdx.x & 31;
  const int nn = lane & 15;
  const int hi = lane >> 4;
  const float bb = bc[n0 + nn];
#pragma unroll
  for (int r = 0; r < 8; ++r) {
    const int b = b0 + r + (hi << 3);
    logits[(size_t)b * D + n0 + nn] = acc[r] + bb;
  }
}

// ---- per-batch softmax over D=512 logits -> packed bf16 comp ------------------
__global__ __launch_bounds__(256) void nse_softmax_comp(
    const float* __restrict__ logits, __bf16* __restrict__ pComp)
{
  __shared__ float red[256];
  const int b = blockIdx.x;
  const int tid = threadIdx.x;
  const float v0 = logits[(size_t)b * D + tid];
  const float v1 = logits[(size_t)b * D + tid + 256];
  red[tid] = fmaxf(v0, v1); __syncthreads();
  for (int off = 128; off > 0; off >>= 1) { if (tid < off) red[tid] = fmaxf(red[tid], red[tid + off]); __syncthreads(); }
  const float mx = red[0];
  __syncthreads();
  const float e0 = __expf(v0 - mx), e1 = __expf(v1 - mx);
  red[tid] = e0 + e1; __syncthreads();
  for (int off = 128; off > 0; off >>= 1) { if (tid < off) red[tid] += red[tid + off]; __syncthreads(); }
  const float inv = 1.0f / red[0];
  pComp[packA_index(b, tid)]       = (__bf16)(e0 * inv);
  pComp[packA_index(b, tid + 256)] = (__bf16)(e1 * inv);
}

// ---- M[b,n,:] = (1 - z[b,n]) * (M[b,n,:] + hw[b,:])  (in place) ---------------
__global__ __launch_bounds__(256) void nse_mem_update(
    float* __restrict__ M, const float* __restrict__ z, const float* __restrict__ hw)
{
  const int i = blockIdx.x * 256 + threadIdx.x;
  const int total = (B * N * D) / 4;
  if (i >= total) return;
  float4* M4 = (float4*)M;
  const int e0 = i * 4;
  const int bn = e0 / D;
  const int d  = e0 % D;
  const int b  = bn >> 9;  // N = 512
  const float er = 1.0f - z[bn];
  float4 mv = M4[i];
  const float4 hv = *(const float4*)(hw + (size_t)b * D + d);
  mv.x = er * (mv.x + hv.x);
  mv.y = er * (mv.y + hv.y);
  mv.z = er * (mv.z + hv.z);
  mv.w = er * (mv.w + hv.w);
  M4[i] = mv;
}

extern "C" void kernel_launch(void* const* d_in, const int* in_sizes, int n_in,
                              void* d_out, int out_size, void* d_ws, size_t ws_size,
                              hipStream_t stream)
{
  (void)in_sizes; (void)n_in; (void)out_size; (void)ws_size;
  const float* emb    = (const float*)d_in[0];
  const float* hr0    = (const float*)d_in[1];
  const float* cr0    = (const float*)d_in[2];
  const float* hw0    = (const float*)d_in[3];
  const float* cw0    = (const float*)d_in[4];
  const float* M0     = (const float*)d_in[5];
  const unsigned char* mask = (const unsigned char*)d_in[6];
  // d_in[7] = M_que (unused)
  const float* W_ih_r = (const float*)d_in[8];
  const float* W_hh_r = (const float*)d_in[9];
  const float* b_r    = (const float*)d_in[10];
  const float* W_ih_w = (const float*)d_in[11];
  const float* W_hh_w = (const float*)d_in[12];
  const float* b_w    = (const float*)d_in[13];
  const float* W_comp = (const float*)d_in[14];
  const float* b_comp = (const float*)d_in[15];

  float* out = (float*)d_out;
  const size_t BD  = (size_t)B * D;
  const size_t BN  = (size_t)B * N;
  const size_t BND = (size_t)B * N * D;
  const size_t WSZ = (size_t)4 * D * D;   // 4D x D weight elements

  // -------- workspace layout: fp32 region, then 32B-aligned bf16 region --------
  float* ws = (float*)d_ws;
  float* Mws    = ws;              ws += BND;
  float* hrb[2] = { ws, ws + BD }; ws += 2 * BD;
  float* crb[2] = { ws, ws + BD }; ws += 2 * BD;
  float* hwb[2] = { ws, ws + BD }; ws += 2 * BD;
  float* cwb[2] = { ws, ws + BD }; ws += 2 * BD;
  float* logits = ws;              ws += BD;
  float* numv   = ws;              ws += BN;
  float* nMv    = ws;              ws += BN;
  float* zv     = ws;              ws += BN;

  __bf16* bw = (__bf16*)ws;
  __bf16* pWihR  = bw; bw += WSZ;
  __bf16* pWhhR  = bw; bw += WSZ;
  __bf16* pWihW  = bw; bw += WSZ;
  __bf16* pWhhW  = bw; bw += WSZ;
  __bf16* pWc    = bw; bw += (size_t)D * 2 * D;
  __bf16* pX     = bw; bw += (size_t)T_STEPS * BD;
  __bf16* pHrI   = bw; bw += BD;
  __bf16* pHwI   = bw; bw += BD;
  __bf16* pHrB[2] = { bw, bw + BD }; bw += 2 * BD;
  __bf16* pHwB[2] = { bw, bw + BD }; bw += 2 * BD;
  __bf16* pMvec  = bw; bw += BD;
  __bf16* pCompv = bw; bw += BD;

  hipMemcpyAsync(Mws, M0, BND * sizeof(float), hipMemcpyDeviceToDevice, stream);

  // -------- one-time packing (weights for whole scan, emb for all T) -----------
  const int wtot = 4 * D * D;
  pack_w<<<(wtot + 255) / 256, 256, 0, stream>>>(W_ih_r, pWihR, 4 * D, D);
  pack_w<<<(wtot + 255) / 256, 256, 0, stream>>>(W_hh_r, pWhhR, 4 * D, D);
  pack_w<<<(wtot + 255) / 256, 256, 0, stream>>>(W_ih_w, pWihW, 4 * D, D);
  pack_w<<<(wtot + 255) / 256, 256, 0, stream>>>(W_hh_w, pWhhW, 4 * D, D);
  const int ctot = D * 2 * D;
  pack_w<<<(ctot + 255) / 256, 256, 0, stream>>>(W_comp, pWc, D, 2 * D);
  const int xtot = T_STEPS * (int)BD;
  pack_a_batch<<<(xtot + 255) / 256, 256, 0, stream>>>(emb, pX, xtot);
  pack_a_batch<<<((int)BD + 255) / 256, 256, 0, stream>>>(hr0, pHrI, (int)BD);
  pack_a_batch<<<((int)BD + 255) / 256, 256, 0, stream>>>(hw0, pHwI, (int)BD);

  // -------- sequential scan ----------------------------------------------------
  const dim3 gGemm(D / 16, B / 16);  // 32 x 2 waves, one 16x16 tile each
  for (int t = 0; t < T_STEPS; ++t) {
    const __bf16* pXt    = pX + (size_t)t * BD;
    const __bf16* pHr_in = (t == 0) ? pHrI : pHrB[(t - 1) & 1];
    const __bf16* pHw_in = (t == 0) ? pHwI : pHwB[(t - 1) & 1];
    const float* cr_in = (t == 0) ? cr0 : crb[(t - 1) & 1];
    const float* cw_in = (t == 0) ? cw0 : cwb[(t - 1) & 1];
    float* hr_o = hrb[t & 1]; float* cr_o = crb[t & 1];
    float* hw_o = hwb[t & 1]; float* cw_o = cwb[t & 1];

    nse_lstm_gates<<<gGemm, 32, 0, stream>>>(pXt, pHr_in, cr_in, pWihR, pWhhR, b_r,
                                             hr_o, cr_o, nullptr, pHrB[t & 1]);
    nse_sim<<<(B * N) / 8, 256, 0, stream>>>(hr_o, Mws, numv, nMv);
    nse_softmax_z<<<B, 256, 0, stream>>>(hr_o, numv, nMv, mask, zv);
    nse_read<<<dim3(D / 256, B), 256, 0, stream>>>(zv, Mws, pMvec);
    nse_comp_gemm<<<gGemm, 32, 0, stream>>>(pHrB[t & 1], pMvec, pWc, b_comp, logits);
    nse_softmax_comp<<<B, 256, 0, stream>>>(logits, pCompv);
    nse_lstm_gates<<<gGemm, 32, 0, stream>>>(pCompv, pHw_in, cw_in, pWihW, pWhhW, b_w,
                                             hw_o, cw_o, out + (size_t)t * BD, pHwB[t & 1]);
    nse_mem_update<<<(B * N * D / 4 + 255) / 256, 256, 0, stream>>>(Mws, zv, hw_o);
  }

  // -------- final states: outputs | hr | cr | hw | cw | M ----------------------
  const int last = (T_STEPS - 1) & 1;
  float* tail = out + (size_t)T_STEPS * BD;
  hipMemcpyAsync(tail,          hrb[last], BD * sizeof(float), hipMemcpyDeviceToDevice, stream);
  hipMemcpyAsync(tail + BD,     crb[last], BD * sizeof(float), hipMemcpyDeviceToDevice, stream);
  hipMemcpyAsync(tail + 2 * BD, hwb[last], BD * sizeof(float), hipMemcpyDeviceToDevice, stream);
  hipMemcpyAsync(tail + 3 * BD, cwb[last], BD * sizeof(float), hipMemcpyDeviceToDevice, stream);
  hipMemcpyAsync(tail + 4 * BD, Mws,      BND * sizeof(float), hipMemcpyDeviceToDevice, stream);
}
